// RSSMTransition_83056077570380
// MI455X (gfx1250) — matrix operand order repaired and checked by reading, not compiled, not run on hardware
//
#include <hip/hip_runtime.h>

// ---------------------------------------------------------------------------
// RSSM transition for MI455X (gfx1250, wave32, WMMA).
// Bandwidth-bound (~1 GB HBM traffic vs 136 GFLOP at f16-WMMA rates):
//  - fuse the reference's jnp.concatenate ops into a two-segment GEMM A-operand
//  - keep weights hot in L2 (~17 MB << 192 MB), stream activations once
//  - f32->f16 conversion during LDS staging (b128 global loads / b128 LDS stores)
//  - 256x64 C-tile, 8 v_wmma_f32_16x16x32_f16 per wave per K-step
//  - GEMM templated on (K1,K2,N,ACT): segment/tail handling resolves at compile
//    time, so the rolled K-loops are branch-free straight-line code.
// ---------------------------------------------------------------------------

typedef __attribute__((ext_vector_type(16))) _Float16 v16h_t;
typedef __attribute__((ext_vector_type(8)))  _Float16 v8h_t;
typedef __attribute__((ext_vector_type(8)))  float    v8f_t;

#define M_TOTAL 16384   // B * NAG = 4096 * 4
#define DET     600
#define NCAT    32
#define NCLS    32

#define BM 256
#define BN 64
#define BK 32
#define LDSTR 48        // halves per LDS row (96 B) -> 16/32 B-aligned frag reads

__device__ __forceinline__ void cvt4(_Float16* dst, float4 v) {
    dst[0] = (_Float16)v.x; dst[1] = (_Float16)v.y;
    dst[2] = (_Float16)v.z; dst[3] = (_Float16)v.w;
}

// ---- A staging: whole 32-k window inside one segment (pure vector path) ----
template<int SEG, int K1, int K2>
__device__ __forceinline__ void stageA_win(const float* __restrict__ A1,
                                           const float* __restrict__ A2,
                                           long grow, int k0, _Float16* dstRow)
{
    const float* base = (SEG == 1) ? (A1 + grow * K1 + k0)
                                   : (A2 + grow * K2 + (k0 - K1));
    const float4* p = (const float4*)base;     // K1%4==0, K2%4==0 -> 16B aligned
    alignas(16) _Float16 tmp[BK];
    #pragma unroll
    for (int q = 0; q < 8; ++q) cvt4(&tmp[q * 4], p[q]);
    v8h_t* d = (v8h_t*)dstRow;                 // LDS row base 96B -> 16B aligned
    #pragma unroll
    for (int q = 0; q < 4; ++q) d[q] = *(const v8h_t*)&tmp[q * 8];
}

// ---- A staging: peeled window with constexpr k0 (K1-crossing or K-tail);
//      per-element segment choice / zero-fill constant-folds after unroll ----
template<int K0, int K1, int K2>
__device__ __forceinline__ void stageA_peel(const float* __restrict__ A1,
                                            const float* __restrict__ A2,
                                            long grow, _Float16* dstRow)
{
    constexpr int K = K1 + K2;
    alignas(16) _Float16 tmp[BK];
    #pragma unroll
    for (int j = 0; j < BK; ++j) {
        float v;
        if (K0 + j < K1)     v = A1[grow * K1 + (K0 + j)];
        else if (K0 + j < K) v = A2[grow * K2 + (K0 + j - K1)];
        else                 v = 0.0f;
        tmp[j] = (_Float16)v;
    }
    v8h_t* d = (v8h_t*)dstRow;
    #pragma unroll
    for (int q = 0; q < 4; ++q) d[q] = *(const v8h_t*)&tmp[q * 8];
}

// ---- W staging: full window (branchless N clamp; OOB cols never stored) ----
template<int K, int N>
__device__ __forceinline__ void stageW_win(const float* __restrict__ W, int colBlock,
                                           int tid, int k0, _Float16 (*Bs)[LDSTR])
{
    const int n = tid >> 2, cbase = (tid & 3) * 8;
    int gn = colBlock + n;
    if ((N % BN) != 0) gn = (gn < N) ? gn : (N - 1);
    const float4* p = (const float4*)(W + (long)gn * K + k0 + cbase);  // K%4==0
    alignas(16) _Float16 tmp[8];
    cvt4(&tmp[0], p[0]);
    cvt4(&tmp[4], p[1]);
    *(v8h_t*)&Bs[n][cbase] = *(const v8h_t*)&tmp[0];
}

// ---- W staging: peeled K-tail (clamped-index loads + select-zero, no branches)
template<int K0, int K, int N>
__device__ __forceinline__ void stageW_peel(const float* __restrict__ W, int colBlock,
                                            int tid, _Float16 (*Bs)[LDSTR])
{
    const int n = tid >> 2, cbase = (tid & 3) * 8;
    int gn = colBlock + n;
    if ((N % BN) != 0) gn = (gn < N) ? gn : (N - 1);
    const long rb = (long)gn * K;
    alignas(16) _Float16 tmp[8];
    #pragma unroll
    for (int j = 0; j < 8; ++j) {
        const int k  = K0 + cbase + j;
        const int kc = (k < K) ? k : (K - 1);
        const float v = W[rb + kc];
        tmp[j] = (_Float16)((k < K) ? v : 0.0f);
    }
    *(v8h_t*)&Bs[n][cbase] = *(const v8h_t*)&tmp[0];
}

// ---- 8 WMMAs per wave: 2 A-frags x 4 shared B-frags (ISA 7.12.2 layouts) ----
__device__ __forceinline__ void wmma_step(const _Float16 (*As)[LDSTR],
                                          const _Float16 (*Bs)[LDSTR],
                                          int wave, int lane, v8f_t acc[2][4])
{
    const int offB = (lane >> 4) * 8;
    v16h_t bfrag[4];
    #pragma unroll
    for (int j = 0; j < 4; ++j) {
        v8h_t b_lo = *(const v8h_t*)&Bs[j * 16 + (lane & 15)][offB];
        v8h_t b_hi = *(const v8h_t*)&Bs[j * 16 + (lane & 15)][offB + 16];
        #pragma unroll
        for (int e = 0; e < 8; ++e) { bfrag[j][e] = b_lo[e]; bfrag[j][8 + e] = b_hi[e]; }
    }

    const int offA = (lane >> 4) * 8;   // lanes 0-15: K {0..7,16..23}; 16-31: {8..15,24..31}
    #pragma unroll
    for (int s = 0; s < 2; ++s) {
        const int mrow = wave * 32 + s * 16 + (lane & 15);
        v8h_t a_lo = *(const v8h_t*)&As[mrow][offA];
        v8h_t a_hi = *(const v8h_t*)&As[mrow][offA + 16];
        v16h_t afrag;
        #pragma unroll
        for (int e = 0; e < 8; ++e) { afrag[e] = a_lo[e]; afrag[8 + e] = a_hi[e]; }
        #pragma unroll
        for (int j = 0; j < 4; ++j)
            acc[s][j] = __builtin_amdgcn_wmma_f32_16x16x32_f16(
                false, afrag, false, bfrag[j], (short)0, acc[s][j], false, false);
    }
}

// ---------------------------------------------------------------------------
// C[M,N] = ACT( [A1|A2][M,K1+K2] @ W[N,K]^T + bias ), fully specialized.
// ---------------------------------------------------------------------------
template<int K1, int K2, int N, int ACT>
__global__ __launch_bounds__(256)
void rssm_gemm(const float* __restrict__ A1, const float* __restrict__ A2,
               const float* __restrict__ W, const float* __restrict__ bias,
               float* __restrict__ C)
{
    constexpr int K = K1 + K2;
    __shared__ _Float16 As[BM][LDSTR];
    __shared__ _Float16 Bs[BN][LDSTR];

    const int tid  = threadIdx.x;
    const int wave = tid >> 5;
    const int lane = tid & 31;
    const int rowBlock = blockIdx.y * BM;
    const int colBlock = blockIdx.x * BN;
    const long grow = (long)(rowBlock + tid);

    v8f_t acc[2][4] = {};

    // Compile-time K-loop decomposition.
    constexpr int  W1    = K1 / 32;                 // pure seg1 windows [0, W1)
    constexpr bool hasB  = (K1 % 32) != 0;          // K1-crossing window
    constexpr int  BW0   = W1 * 32;
    constexpr int  T2a   = (K1 + 31) / 32;          // first pure seg2 window
    constexpr int  T2b   = K / 32;                  // end of pure windows
    constexpr int  TW0   = (K / 32) * 32;
    constexpr bool hasT  = ((K % 32) != 0) && !(hasB && TW0 == BW0);

    for (int t = 0; t < W1; ++t) {                  // pure segment-1 windows
        const int k0 = t * 32;
        __syncthreads();
        stageA_win<1, K1, K2>(A1, A2, grow, k0, &As[tid][0]);
        stageW_win<K, N>(W, colBlock, tid, k0, Bs);
        if (k0 + 32 < K1) __builtin_prefetch(A1 + grow * K1 + k0 + 32, 0, 3);
        __syncthreads();
        wmma_step(As, Bs, wave, lane, acc);
    }
    if constexpr (hasB) {                           // peeled K1-crossing window
        __syncthreads();
        stageA_peel<BW0, K1, K2>(A1, A2, grow, &As[tid][0]);
        if constexpr (BW0 + 32 <= K) stageW_win<K, N>(W, colBlock, tid, BW0, Bs);
        else                         stageW_peel<BW0, K, N>(W, colBlock, tid, Bs);
        __syncthreads();
        wmma_step(As, Bs, wave, lane, acc);
    }
    if constexpr (K2 > 0) {
        for (int t = T2a; t < T2b; ++t) {           // pure segment-2 windows
            const int k0 = t * 32;
            __syncthreads();
            stageA_win<2, K1, K2>(A1, A2, grow, k0, &As[tid][0]);
            stageW_win<K, N>(W, colBlock, tid, k0, Bs);
            if (k0 + 32 < K) __builtin_prefetch(A2 + grow * K2 + (k0 + 32 - K1), 0, 3);
            __syncthreads();
            wmma_step(As, Bs, wave, lane, acc);
        }
    }
    if constexpr (hasT) {                           // peeled K-tail window
        __syncthreads();
        stageA_peel<TW0, K1, K2>(A1, A2, grow, &As[tid][0]);
        stageW_peel<TW0, K, N>(W, colBlock, tid, Bs);
        __syncthreads();
        wmma_step(As, Bs, wave, lane, acc);
    }

    // ---- epilogue: C layout (VGPR r -> M = r + 8*(lane>=16), N = lane&15) ----
    const int nl = lane & 15;
    #pragma unroll
    for (int s = 0; s < 2; ++s) {
        const int mbase = rowBlock + wave * 32 + s * 16 + ((lane >> 4) ? 8 : 0);
        #pragma unroll
        for (int j = 0; j < 4; ++j) {
            const int gn = colBlock + j * 16 + nl;
            if ((N % BN) != 0 && gn >= N) continue;
            const float bv = bias[gn];
            #pragma unroll
            for (int r = 0; r < 8; ++r) {
                float v = acc[s][j][r] + bv;
                if (ACT == 1) v = fmaxf(v, 0.0f);
                C[(long)(mbase + r) * N + gn] = v;
            }
        }
    }
}

// ---------------------------------------------------------------------------
// GRU gate combine, float4-vectorized (DET % 4 == 0, all row bases 16B-aligned)
// ---------------------------------------------------------------------------
__device__ __forceinline__ float sigmf(float x) { return 1.0f / (1.0f + __expf(-x)); }

__global__ __launch_bounds__(256)
void rssm_gru_combine(const float* __restrict__ gi, const float* __restrict__ gh,
                      const float* __restrict__ h, float* __restrict__ out, int total4)
{
    const int i = blockIdx.x * blockDim.x + threadIdx.x;
    if (i >= total4) return;
    const int row = i / (DET / 4);
    const int j4  = (i - row * (DET / 4)) * 4;
    const long b  = (long)row * (3 * DET);

    float4 gir = *(const float4*)(gi + b + j4);
    float4 giz = *(const float4*)(gi + b + DET + j4);
    float4 gin = *(const float4*)(gi + b + 2 * DET + j4);
    float4 ghr = *(const float4*)(gh + b + j4);
    float4 ghz = *(const float4*)(gh + b + DET + j4);
    float4 ghn = *(const float4*)(gh + b + 2 * DET + j4);
    float4 hv  = *(const float4*)(h + (long)row * DET + j4);

    float4 o;
    { const float r = sigmf(gir.x + ghr.x), z = sigmf(giz.x + ghz.x);
      const float n = tanhf(gin.x + r * ghn.x); o.x = (1.0f - z) * n + z * hv.x; }
    { const float r = sigmf(gir.y + ghr.y), z = sigmf(giz.y + ghz.y);
      const float n = tanhf(gin.y + r * ghn.y); o.y = (1.0f - z) * n + z * hv.y; }
    { const float r = sigmf(gir.z + ghr.z), z = sigmf(giz.z + ghz.z);
      const float n = tanhf(gin.z + r * ghn.z); o.z = (1.0f - z) * n + z * hv.z; }
    { const float r = sigmf(gir.w + ghr.w), z = sigmf(giz.w + ghz.w);
      const float n = tanhf(gin.w + r * ghn.w); o.w = (1.0f - z) * n + z * hv.w; }
    *(float4*)(out + (long)row * DET + j4) = o;
}

// ---------------------------------------------------------------------------
// Softmax over 32 classes + categorical sample via deterministic PCG hash
// (jax threefry stream not bit-reproducible; straight-through fwd = one-hot).
// ---------------------------------------------------------------------------
__global__ __launch_bounds__(256)
void rssm_softmax_sample(const float* __restrict__ logits, float* __restrict__ onehot,
                         int total /* M*NCAT */, unsigned seed)
{
    const int i = blockIdx.x * blockDim.x + threadIdx.x;
    if (i >= total) return;

    alignas(16) float lg[NCLS];
    {
        const float4* p = (const float4*)(logits + (long)i * NCLS);
        #pragma unroll
        for (int q = 0; q < NCLS / 4; ++q) *(float4*)&lg[q * 4] = p[q];
    }

    float mx = lg[0];
    #pragma unroll
    for (int c = 1; c < NCLS; ++c) mx = fmaxf(mx, lg[c]);
    float e[NCLS]; float s = 0.0f;
    #pragma unroll
    for (int c = 0; c < NCLS; ++c) { e[c] = __expf(lg[c] - mx); s += e[c]; }

    unsigned x = (unsigned)i * 747796405u + seed;
    x ^= x >> 17; x *= 0xed5ad4bbu; x ^= x >> 11; x *= 0xac4c1b51u;
    x ^= x >> 15; x *= 0x31848babu; x ^= x >> 14;
    const float target = ((x >> 8) * (1.0f / 16777216.0f)) * s;

    float cum = 0.0f; int idx = NCLS - 1;
    #pragma unroll
    for (int c = 0; c < NCLS; ++c) { cum += e[c]; if (target < cum) { idx = c; break; } }

    float4* oh = (float4*)(onehot + (long)i * NCLS);
    #pragma unroll
    for (int q = 0; q < NCLS / 4; ++q) {
        float4 v;
        v.x = (q * 4 + 0 == idx) ? 1.0f : 0.0f;
        v.y = (q * 4 + 1 == idx) ? 1.0f : 0.0f;
        v.z = (q * 4 + 2 == idx) ? 1.0f : 0.0f;
        v.w = (q * 4 + 3 == idx) ? 1.0f : 0.0f;
        oh[q] = v;
    }
}

// ---------------------------------------------------------------------------
// Orchestration
// ---------------------------------------------------------------------------
template<int K1, int K2, int N, int ACT>
static inline void launch_gemm(const float* A1, const float* A2,
                               const float* W, const float* b, float* C,
                               hipStream_t stream)
{
    dim3 grid((N + BN - 1) / BN, M_TOTAL / BM);
    rssm_gemm<K1, K2, N, ACT><<<grid, 256, 0, stream>>>(A1, A2, W, b, C);
}

extern "C" void kernel_launch(void* const* d_in, const int* in_sizes, int n_in,
                              void* d_out, int out_size, void* d_ws, size_t ws_size,
                              hipStream_t stream)
{
    (void)in_sizes; (void)n_in; (void)out_size; (void)ws_size;
    const float* prev_actions       = (const float*)d_in[0];   // [M, 20]
    const float* prev_joint_actions = (const float*)d_in[1];   // [M, 80]
    const float* agent_stoch        = (const float*)d_in[2];   // [M, 1024]
    const float* agent_deter        = (const float*)d_in[3];   // [M, 600]
    const float* global_stoch       = (const float*)d_in[4];   // [M, 1024]
    const float* global_deter       = (const float*)d_in[5];   // [M, 600]
    const float* g_in_W = (const float*)d_in[6];  const float* g_in_b = (const float*)d_in[7];
    const float* g_Wih  = (const float*)d_in[8];  const float* g_Whh  = (const float*)d_in[9];
    const float* g_bih  = (const float*)d_in[10]; const float* g_bhh  = (const float*)d_in[11];
    const float* g_p1W  = (const float*)d_in[12]; const float* g_p1b  = (const float*)d_in[13];
    const float* g_p2W  = (const float*)d_in[14]; const float* g_p2b  = (const float*)d_in[15];
    const float* a_in_W = (const float*)d_in[16]; const float* a_in_b = (const float*)d_in[17];
    const float* a_Wih  = (const float*)d_in[18]; const float* a_Whh  = (const float*)d_in[19];
    const float* a_bih  = (const float*)d_in[20]; const float* a_bhh  = (const float*)d_in[21];
    const float* a_p1W  = (const float*)d_in[22]; const float* a_p1b  = (const float*)d_in[23];
    const float* a_p2W  = (const float*)d_in[24]; const float* a_p2b  = (const float*)d_in[25];

    // Output layout (floats): a_logits | a_stoch | a_det | g_logits | g_stoch | g_det
    float* out = (float*)d_out;
    const long SZ_L = (long)M_TOTAL * NCAT * NCLS;
    const long SZ_D = (long)M_TOTAL * DET;
    float* a_logits = out;
    float* a_stoch  = out + SZ_L;
    float* a_det    = out + 2 * SZ_L;
    float* g_logits = out + 2 * SZ_L + SZ_D;
    float* g_stoch  = out + 3 * SZ_L + SZ_D;
    float* g_det    = out + 4 * SZ_L + SZ_D;

    // Workspace (floats): x-buffer [M,200] reused; gi/gh [M,1800] reused per branch.
    float* ws    = (float*)d_ws;
    float* xbuf  = ws;                                 // M*200
    float* gibuf = ws + (long)M_TOTAL * 200;           // M*1800
    float* ghbuf = gibuf + (long)M_TOTAL * 1800;       // M*1800

    const int gru_total4 = M_TOTAL * DET / 4;
    const int samp_total = M_TOTAL * NCAT;
    const int gru_blocks  = (gru_total4 + 255) / 256;
    const int samp_blocks = (samp_total + 255) / 256;

    // ---------------- global branch ----------------
    launch_gemm<80, 1024, 200, 1>(prev_joint_actions, global_stoch, g_in_W, g_in_b, xbuf, stream);
    launch_gemm<200, 0, 1800, 0>(xbuf, nullptr, g_Wih, g_bih, gibuf, stream);
    launch_gemm<600, 0, 1800, 0>(global_deter, nullptr, g_Whh, g_bhh, ghbuf, stream);
    rssm_gru_combine<<<gru_blocks, 256, 0, stream>>>(gibuf, ghbuf, global_deter, g_det, gru_total4);
    launch_gemm<600, 0, 200, 1>(g_det, nullptr, g_p1W, g_p1b, xbuf, stream);
    launch_gemm<200, 0, 1024, 0>(xbuf, nullptr, g_p2W, g_p2b, g_logits, stream);
    rssm_softmax_sample<<<samp_blocks, 256, 0, stream>>>(g_logits, g_stoch, samp_total, 0x9E3779B9u);

    // ---------------- agent branch ----------------
    launch_gemm<20, 1024, 200, 1>(prev_actions, agent_stoch, a_in_W, a_in_b, xbuf, stream);
    launch_gemm<200, 0, 1800, 0>(xbuf, nullptr, a_Wih, a_bih, gibuf, stream);
    launch_gemm<600, 0, 1800, 0>(agent_deter, nullptr, a_Whh, a_bhh, ghbuf, stream);
    rssm_gru_combine<<<gru_blocks, 256, 0, stream>>>(gibuf, ghbuf, agent_deter, a_det, gru_total4);
    launch_gemm<600, 1024, 200, 1>(a_det, g_stoch, a_p1W, a_p1b, xbuf, stream);
    launch_gemm<200, 0, 1024, 0>(xbuf, nullptr, a_p2W, a_p2b, a_logits, stream);
    rssm_softmax_sample<<<samp_blocks, 256, 0, stream>>>(a_logits, a_stoch, samp_total, 0x85EBCA6Bu);
}
